// FlashAttention_22436909154623
// MI455X (gfx1250) — matrix operand order repaired and checked by reading, not compiled, not run on hardware
//
#include <hip/hip_runtime.h>

typedef __attribute__((ext_vector_type(16))) _Float16 v16h;
typedef __attribute__((ext_vector_type(8)))  _Float16 v8h;
typedef __attribute__((ext_vector_type(2)))  _Float16 v2h;
typedef __attribute__((ext_vector_type(8)))  float    v8f;

#define MASKF (-30000.0f)   // f16-exact "minus infinity" (survives f16 P conversion)

constexpr int B = 4, T = 1024, S = 4096, H = 16, D = 64;
constexpr int CHUNK  = 64;          // key/value tile (S direction)
constexpr int QTILE  = 128;         // query rows per workgroup (16 per wave, 8 waves)
constexpr int WAVES  = QTILE / 16;
constexpr int THREADS = WAVES * 32; // 256
constexpr int LDK    = 72;          // LDS row stride in halves: 144B -> 16B aligned, bank-skewed
constexpr int NCH    = S / CHUNK;
constexpr int KVROW  = 2 * H * D;   // kv stride between s rows (floats)
constexpr int HD     = H * D;
constexpr int VROWS  = D + 16;      // extra 16 rows: row 64 = ones column -> l rides in accumulator

// concat two contiguous 8-half LDS loads (each lowers to ds_load_b128) into a v16h fragment
__device__ __forceinline__ v16h ld2x8(const _Float16* lo, const _Float16* hi) {
  v8h a = *(const v8h*)lo;
  v8h b = *(const v8h*)hi;
  return __builtin_shufflevector(a, b, 0,1,2,3,4,5,6,7,8,9,10,11,12,13,14,15);
}

__device__ __forceinline__ v8h pack8(const float4& a, const float4& b) {
  v8h r;
  r[0] = (_Float16)a.x; r[1] = (_Float16)a.y; r[2] = (_Float16)a.z; r[3] = (_Float16)a.w;
  r[4] = (_Float16)b.x; r[5] = (_Float16)b.y; r[6] = (_Float16)b.z; r[7] = (_Float16)b.w;
  return r;
}

__global__ __launch_bounds__(THREADS) void fa_fwd_wmma(
    const float* __restrict__ q, const float* __restrict__ kv,
    const unsigned char* __restrict__ kpm, const int* __restrict__ causalp,
    float* __restrict__ out) {
  __shared__ _Float16 Klds[2][CHUNK * LDK];     // [s][d]  f16
  __shared__ _Float16 Vlds[2][VROWS * LDK];     // [d][s]  f16 transposed; rows >= D are the ones tile
  __shared__ _Float16 Plds[WAVES][16 * LDK];    // per-wave P (C-layout -> A-layout) staging
  __shared__ float    Mlds[2][CHUNK];           // staged additive padding-mask bias {0, MASKF}

  const int tid  = threadIdx.x;
  const int wave = tid >> 5;
  const int lane = tid & 31;
  const int g    = lane >> 4;   // half-wave group (K-chunk / row-half selector)
  const int ln   = lane & 15;   // N column / A row within 16x16 tile

  const int tTiles = T / QTILE;
  const int tTile  = blockIdx.x % tTiles;
  const int hh     = (blockIdx.x / tTiles) % H;
  const int b      =  blockIdx.x / (tTiles * H);
  const int tBase  = tTile * QTILE;

  const int causal = causalp[0];
  const int diag   = S - T;
  // log2-domain softmax: fold log2(e)/sqrt(D) into the score scale so every
  // exponential is a bare v_exp_f32 (TRANS pipe, co-executes with WMMA).
  const float scl = 0.18033688011112042f;  // 0.125 * log2(e)

  // ---- ones tile for V (rows 64..79): row 64 = 1.0, rest 0.0; written once ----
  for (int i = tid; i < 16 * CHUNK; i += THREADS) {
    const int rr = D + (i >> 6);
    const int ss = i & (CHUNK - 1);
    const _Float16 val = (rr == D) ? (_Float16)1.0f : (_Float16)0.0f;
    Vlds[0][rr * LDK + ss] = val;
    Vlds[1][rr * LDK + ss] = val;
  }

  // ---- Q A-fragments (16x32 per k-step), loaded once from global in A VGPR layout ----
  const int qRow = tBase + wave * 16 + ln;
  const float* qp = q + (((size_t)b * T + qRow) * H + hh) * D;
  v16h qa[2];
  #pragma unroll
  for (int ks = 0; ks < 2; ++ks) {
    const int d0 = ks * 32 + g * 16;
    #pragma unroll
    for (int c = 0; c < 8; ++c) {
      qa[ks][c]     = (_Float16)qp[d0 + c];        // K = ks*32 + g*16 + c
      qa[ks][c + 8] = (_Float16)qp[d0 + 8 + c];    // K = ks*32 + g*16 + 8 + c
    }
  }

  v8f   o[5] = {};   // 4 N-tiles over D + tile 4 (col 0 = running softmax denominator l)
  float m_r[8];      // per-row running max (log2 domain), rows m = g*8 + r
  #pragma unroll
  for (int r = 0; r < 8; ++r) m_r[r] = -3.0e38f;

  const size_t kvB = (size_t)b * S * KVROW + (size_t)hh * D;
  const int srow = tid >> 2;          // staging: row of the 64-row chunk (256 thr / 64 rows)
  const int dh   = (tid & 3) * 16;    // staging: 16-float slice of the 64-wide D dim
  const int mcol = tid & (CHUNK - 1); // branchless mask staging column (4x redundant)

  float4 kreg[4], vreg[4];
  float  mreg = 0.f;

  auto issueK = [&](int sBase) {
    const float* kp = kv + kvB + (size_t)(sBase + srow) * KVROW + dh;
    #pragma unroll
    for (int i = 0; i < 4; ++i) kreg[i] = *(const float4*)(kp + i * 4);
    mreg = kpm[(size_t)b * S + sBase + mcol] ? 0.f : MASKF;
  };
  auto issueV = [&](int sBase) {
    const float* vp = kv + kvB + (size_t)(sBase + srow) * KVROW + HD + dh;
    #pragma unroll
    for (int i = 0; i < 4; ++i) vreg[i] = *(const float4*)(vp + i * 4);
  };
  auto stageK = [&](int buf) {  // K: packed f16, two b128 stores per thread
    _Float16* kd = &Klds[buf][srow * LDK + dh];
    *(v8h*)(kd)     = pack8(kreg[0], kreg[1]);
    *(v8h*)(kd + 8) = pack8(kreg[2], kreg[3]);
    Mlds[buf][mcol] = mreg;
  };
  auto stageV = [&](int buf) {  // V: scatter-transposed scalar stores (must land as [d][s])
    #pragma unroll
    for (int i = 0; i < 4; ++i) {
      const int d0 = dh + i * 4;
      Vlds[buf][(d0+0)*LDK + srow] = (_Float16)vreg[i].x;
      Vlds[buf][(d0+1)*LDK + srow] = (_Float16)vreg[i].y;
      Vlds[buf][(d0+2)*LDK + srow] = (_Float16)vreg[i].z;
      Vlds[buf][(d0+3)*LDK + srow] = (_Float16)vreg[i].w;
    }
  };

  issueK(0); issueV(0);
  stageK(0); stageV(0);
  __syncthreads();

  for (int cs = 0; cs < NCH; ++cs) {
    const int buf   = cs & 1;
    const int sBase = cs * CHUNK;
    // causal: chunks entirely above the band contribute ~0 -> uniform early exit
    if (causal && sBase > tBase + QTILE - 1 + diag) break;
    const bool more = (cs + 1 < NCH);

    if (cs + 2 < NCH) {  // L2 prefetch two chunks ahead -> global_prefetch_b8
      const float* pf = kv + kvB + (size_t)(sBase + 2 * CHUNK + srow) * KVROW + dh;
      __builtin_prefetch(pf, 0, 0);
      __builtin_prefetch(pf + HD, 0, 0);
    }
    if (more) issueK(sBase + CHUNK);   // K loads in flight during QK

    // ---- scores S = Q*K^T : fragments batch-preloaded into distinct registers so
    //      the k-step-1 ds_loads fly while the k-step-0 WMMAs execute ----
    v8f sc[4];
    {
      v16h kf[4];
      #pragma unroll
      for (int nt = 0; nt < 4; ++nt) {
        const _Float16* kb = &Klds[buf][(nt * 16 + ln) * LDK + g * 16];
        kf[nt] = ld2x8(kb, kb + 8);
      }
      #pragma unroll
      for (int nt = 0; nt < 4; ++nt) {
        v8f zz = {};
        sc[nt] = __builtin_amdgcn_wmma_f32_16x16x32_f16(false, qa[0], false, kf[nt], (short)0, zz, false, false);
      }
      #pragma unroll
      for (int nt = 0; nt < 4; ++nt) {
        const _Float16* kb = &Klds[buf][(nt * 16 + ln) * LDK + 32 + g * 16];
        kf[nt] = ld2x8(kb, kb + 8);
      }
      #pragma unroll
      for (int nt = 0; nt < 4; ++nt)
        sc[nt] = __builtin_amdgcn_wmma_f32_16x16x32_f16(false, qa[1], false, kf[nt], (short)0, sc[nt], false, false);
    }

    if (more) { stageK(buf ^ 1); issueV(sBase + CHUNK); }  // free kreg early; V latency covered below

    // ---- additive padding-mask bias + log2-domain scale: one FMA per element ----
    float madd[4];
    #pragma unroll
    for (int nt = 0; nt < 4; ++nt) madd[nt] = Mlds[buf][nt * 16 + ln];
    #pragma unroll
    for (int nt = 0; nt < 4; ++nt)
      #pragma unroll
      for (int r = 0; r < 8; ++r)
        sc[nt][r] = __builtin_fmaf(sc[nt][r], scl, madd[nt]);

    // causal band: uniform scalar branch, zero cost when causal==0
    const bool needC = causal && (sBase + CHUNK - 1 > tBase + diag);
    if (needC) {
      #pragma unroll
      for (int nt = 0; nt < 4; ++nt) {
        const int sAbs = sBase + nt * 16 + ln;
        #pragma unroll
        for (int r = 0; r < 8; ++r) {
          const int tAbs = tBase + wave * 16 + g * 8 + r;
          if (sAbs > tAbs + diag) sc[nt][r] = MASKF;
        }
      }
    }

    // ---- packed-f16 row-max reduction (2 rows per shuffle) across 16 lanes ----
    v2h pm[4];
    #pragma unroll
    for (int i = 0; i < 4; ++i) {
      const int r0 = 2 * i, r1 = 2 * i + 1;
      const float a = fmaxf(fmaxf(sc[0][r0], sc[1][r0]), fmaxf(sc[2][r0], sc[3][r0]));
      const float c = fmaxf(fmaxf(sc[0][r1], sc[1][r1]), fmaxf(sc[2][r1], sc[3][r1]));
      pm[i][0] = (_Float16)a; pm[i][1] = (_Float16)c;
    }
    #pragma unroll
    for (int ms = 1; ms < 16; ms <<= 1) {
      #pragma unroll
      for (int i = 0; i < 4; ++i) {
        const int iy = __shfl_xor(__builtin_bit_cast(int, pm[i]), ms, 32);
        pm[i] = __builtin_elementwise_max(pm[i], __builtin_bit_cast(v2h, iy));
      }
    }

    // ---- online update: p = exp2(x - mn); alpha computed per row ----
    float al[8];
    bool  chg = false;
    #pragma unroll
    for (int r = 0; r < 8; ++r) {
      const float rm = (float)pm[r >> 1][r & 1];
      const float mn = fmaxf(m_r[r], rm);
      al[r] = __builtin_amdgcn_exp2f(m_r[r] - mn);
      chg  |= (al[r] != 1.0f);
      m_r[r] = mn;
      #pragma unroll
      for (int nt = 0; nt < 4; ++nt)
        sc[nt][r] = __builtin_amdgcn_exp2f(sc[nt][r] - mn);
    }

    // ---- P stores first (overlap ds latency with the conditional rescale) ----
    _Float16* pw = &Plds[wave][0];
    #pragma unroll
    for (int nt = 0; nt < 4; ++nt)
      #pragma unroll
      for (int r = 0; r < 8; ++r)
        pw[(g * 8 + r) * LDK + nt * 16 + ln] = (_Float16)sc[nt][r];

    // wave-uniform skip: rows' maxes unchanged -> all alphas are exactly 1.0
    if (__any(chg)) {
      #pragma unroll
      for (int r = 0; r < 8; ++r)
        #pragma unroll
        for (int nt = 0; nt < 5; ++nt) o[nt][r] *= al[r];
    }

    asm volatile("s_wait_dscnt 0" ::: "memory");

    v16h pa[2];
    #pragma unroll
    for (int ks = 0; ks < 2; ++ks) {
      const int c0 = ks * 32 + g * 16;
      pa[ks] = ld2x8(&pw[ln * LDK + c0], &pw[ln * LDK + c0 + 8]);
    }

    // ---- O += P*V (5th tile accumulates row-sums via the ones column);
    //      all 5 fragments of a k-step preloaded into distinct registers ----
    #pragma unroll
    for (int ks = 0; ks < 2; ++ks) {
      v16h vf[5];
      #pragma unroll
      for (int nt = 0; nt < 5; ++nt) {
        const _Float16* vb = &Vlds[buf][(nt * 16 + ln) * LDK + ks * 32 + g * 16];
        vf[nt] = ld2x8(vb, vb + 8);
      }
      #pragma unroll
      for (int nt = 0; nt < 5; ++nt)
        o[nt] = __builtin_amdgcn_wmma_f32_16x16x32_f16(false, pa[ks], false, vf[nt], (short)0, o[nt], false, false);
    }

    if (more) stageV(buf ^ 1);
    __syncthreads();
  }

  // ---- normalize by l (column 0 of tile 4, broadcast from lane g*16); NT stores ----
  #pragma unroll
  for (int r = 0; r < 8; ++r) {
    const float l = __shfl(o[4][r], g * 16, 32);
    const float inv = __builtin_amdgcn_rcpf(l);
    const int t = tBase + wave * 16 + g * 8 + r;
    #pragma unroll
    for (int nt = 0; nt < 4; ++nt)
      __builtin_nontemporal_store(o[nt][r] * inv,
          &out[(((size_t)b * T + t) * H + hh) * D + nt * 16 + ln]);
  }
}

extern "C" void kernel_launch(void* const* d_in, const int* in_sizes, int n_in,
                              void* d_out, int out_size, void* d_ws, size_t ws_size,
                              hipStream_t stream) {
  (void)in_sizes; (void)n_in; (void)out_size; (void)d_ws; (void)ws_size;
  const float*         q    = (const float*)d_in[0];
  const float*         kv   = (const float*)d_in[1];
  const unsigned char* mask = (const unsigned char*)d_in[2];
  const int*           cz   = (const int*)d_in[3];
  float*               out  = (float*)d_out;

  dim3 grid(B * H * (T / QTILE));   // 512 blocks
  dim3 block(THREADS);              // 8 wave32 waves per block
  fa_fwd_wmma<<<grid, block, 0, stream>>>(q, kv, mask, cz, out);
}
